// TopK_SparseMultiHeadAttention_46325517255238
// MI455X (gfx1250) — matrix operand order repaired and checked by reading, not compiled, not run on hardware
//
#include <hip/hip_runtime.h>
#include <cstdint>

#define B_    8
#define S_    1024
#define DM_   1024
#define H_    16
#define DK_   64
#define TOPK_ 32

typedef __attribute__((ext_vector_type(16))) __bf16 v16bf;
typedef __attribute__((ext_vector_type(8)))  float  v8f;

union BF16Frag {
    v16bf v;
    unsigned short s[16];
    uint4 q[2];
};

__device__ __forceinline__ unsigned short f2bf(float f) {
    unsigned int u = __float_as_uint(f);
    u += 0x7FFFu + ((u >> 16) & 1u);          // round-to-nearest-even
    return (unsigned short)(u >> 16);
}

__device__ __forceinline__ void wait_async0() {
#if __has_builtin(__builtin_amdgcn_s_wait_asynccnt)
    __builtin_amdgcn_s_wait_asynccnt(0);
#else
    asm volatile("s_wait_asynccnt 0x0" ::: "memory");
#endif
}

// ---------------------------------------------------------------------------
// Projection GEMM: out[b,h,s,d] = X[m,:] @ W[:,n] + bias[n]
// X: [8192,1024] f32, W: [1024,1024] f32 (row-major [k][n]).
// Block = 8 waves; block computes a 16(M) x 128(N) tile, each wave 16x16.
// W k-slab DMA'd into LDS via GLOBAL_LOAD_ASYNC_TO_LDS_B128 (ASYNCcnt),
// converted to bf16 at fragment-build time; WMMA bf16 K=32 steps, f32 acc.
// ---------------------------------------------------------------------------
__global__ void __launch_bounds__(256)
proj_gemm_kernel(const float* __restrict__ X, const float* __restrict__ W,
                 const float* __restrict__ bias,
                 unsigned short* __restrict__ out_bf,
                 float* __restrict__ out_f, int store_f32)
{
    __shared__ float wlf[32 * 132];   // 32 k-rows x 128 cols f32, pitch 132

    const int bn      = blockIdx.x & 7;
    const int bm      = blockIdx.x >> 3;
    const int row0    = bm << 4;
    const int colbase = bn << 7;
    const int wv      = threadIdx.x >> 5;
    const int lane    = threadIdx.x & 31;
    const int nl      = lane & 15;
    const int hh      = lane >> 4;

    // per-thread async-copy slice: 16 consecutive floats (64B = 4 x b128)
    const int r  = threadIdx.x >> 3;          // 0..31  (k row within slab)
    const int c0 = (threadIdx.x & 7) << 4;    // 0..112 (col within 128)
    const unsigned ldsbase = (unsigned)(uintptr_t)(&wlf[r * 132 + c0]);
    const float* wp0 = W + (size_t)r * DM_ + colbase + c0;

    v8f acc = {};
    for (int kk = 0; kk < DM_; kk += 32) {
        __syncthreads();                       // prior readers done
        const float* wp = wp0 + (size_t)kk * DM_;
        #pragma unroll
        for (int j = 0; j < 4; ++j) {
            const unsigned long long ga = (unsigned long long)(uintptr_t)(wp + j * 4);
            asm volatile("global_load_async_to_lds_b128 %0, %1, off"
                         :: "v"(ldsbase + j * 16), "v"(ga) : "memory");
        }
        if (kk + 32 < DM_)
            __builtin_prefetch(wp + (size_t)32 * DM_, 0, 1);   // next slab
        wait_async0();
        __syncthreads();                       // slab visible to all waves

        // A fragment (16x32 bf16, ISA 16-bit A layout): lane: M=lane%16,
        // K chunks {kk+h*8..+7} and {kk+16+h*8..+7}
        BF16Frag a;
        const float* xr = X + (size_t)(row0 + nl) * DM_ + kk;
        #pragma unroll
        for (int j = 0; j < 8; ++j) a.s[j]     = f2bf(xr[hh * 8 + j]);
        #pragma unroll
        for (int j = 0; j < 8; ++j) a.s[8 + j] = f2bf(xr[16 + hh * 8 + j]);

        // B fragment (32x16 bf16): lane: N=lane%16, K chunks {h*8..+7} and {16+h*8..+7}
        BF16Frag bfr;
        #pragma unroll
        for (int j = 0; j < 16; ++j)
            bfr.s[j] = f2bf(wlf[((j & 7) + hh * 8 + ((j >> 3) << 4)) * 132 + (wv << 4) + nl]);

        acc = __builtin_amdgcn_wmma_f32_16x16x32_bf16(
            false, a.v, false, bfr.v, (short)0, acc, false, false);
    }

    // C/D layout: lane L, vgpr r -> M = r + (L/16)*8, N = L%16
    const int n  = colbase + (wv << 4) + nl;
    const int hd = n >> 6, d = n & 63;
    const float bsv = bias[n];
    #pragma unroll
    for (int rr = 0; rr < 8; ++rr) {
        const int m  = row0 + rr + hh * 8;
        const int bb = m >> 10, ss = m & 1023;
        const size_t oidx = (((size_t)(bb * H_ + hd)) * S_ + ss) * DK_ + d;
        const float val = acc[rr] + bsv;
        if (store_f32) out_f[oidx] = val;
        else           out_bf[oidx] = f2bf(val);
    }
}

// ---------------------------------------------------------------------------
// V_sum[b,h,d] = sum_s v[b,h,s,d]
// ---------------------------------------------------------------------------
__global__ void __launch_bounds__(256)
vsum_kernel(const float* __restrict__ vf, float* __restrict__ vsum)
{
    const int idx = blockIdx.x * blockDim.x + threadIdx.x;   // 0..8191
    const int bh = idx >> 6, d = idx & 63;
    const float* p = vf + (size_t)bh * S_ * DK_ + d;
    float s = 0.f;
    for (int k = 0; k < S_; ++k) s += p[(size_t)k * DK_];
    vsum[idx] = s;
}

// ---------------------------------------------------------------------------
// Fused scores + exact top-32 + context.
// One block per (b, h, 16-query tile). 8 waves.
// Phase 1: 16x1024 score strip via WMMA bf16 -> LDS (f32, scaled by 1/8).
// Phase 2: per row, 32x exact argmax extraction (lane scan + shfl_xor
// reduction), accumulate (e^s - 1) weights against v, add V_sum correction.
// ---------------------------------------------------------------------------
__global__ void __launch_bounds__(256)
topk_attn_kernel(const unsigned short* __restrict__ qbf,
                 const unsigned short* __restrict__ kbf,
                 const float* __restrict__ vf,
                 const float* __restrict__ vsum,
                 float* __restrict__ out)
{
    __shared__ float sc[16 * S_];             // 64 KB score strip

    const int bh = blockIdx.x >> 6;           // 0..127
    const int qt = blockIdx.x & 63;
    const int s0 = qt << 4;
    const int b  = bh >> 4;
    const int h  = bh & 15;

    const int wv   = threadIdx.x >> 5;
    const int lane = threadIdx.x & 31;
    const int nl   = lane & 15;
    const int hh   = lane >> 4;

    // q fragments (A, 16x64 over two K=32 steps) loaded once, reused 8x
    const unsigned short* qrow = qbf + (((size_t)bh * S_) + (s0 + nl)) * DK_;
    BF16Frag a0, a1;
    a0.q[0] = *(const uint4*)(qrow +  0 + hh * 8);
    a0.q[1] = *(const uint4*)(qrow + 16 + hh * 8);
    a1.q[0] = *(const uint4*)(qrow + 32 + hh * 8);
    a1.q[1] = *(const uint4*)(qrow + 48 + hh * 8);

    for (int kt = wv; kt < S_ / 16; kt += 8) {
        const int j0 = kt << 4;
        // B fragment: scores = q . k^T, so B[d][j] = k[j][d]; per lane the
        // 16 needed d-values are two 8-runs (d = 8h.., 16+8h..) of k[b,h,j,:]  -> two 16B loads.
        const unsigned short* krow = kbf + (((size_t)bh * S_) + (j0 + nl)) * DK_;
        if (kt + 8 < S_ / 16)
            __builtin_prefetch(krow + 8 * 16 * DK_, 0, 1);     // next key tile
        BF16Frag b0, b1;
        b0.q[0] = *(const uint4*)(krow + hh * 8);
        b0.q[1] = *(const uint4*)(krow + 16 + hh * 8);
        b1.q[0] = *(const uint4*)(krow + 32 + hh * 8);
        b1.q[1] = *(const uint4*)(krow + 48 + hh * 8);

        v8f c = {};
        c = __builtin_amdgcn_wmma_f32_16x16x32_bf16(
            false, a0.v, false, b0.v, (short)0, c, false, false);
        c = __builtin_amdgcn_wmma_f32_16x16x32_bf16(
            false, a1.v, false, b1.v, (short)0, c, false, false);

        #pragma unroll
        for (int rr = 0; rr < 8; ++rr)
            sc[(rr + hh * 8) * S_ + j0 + nl] = c[rr] * 0.125f;  // 1/sqrt(64)
    }
    __syncthreads();

    const float* vbase = vf + (size_t)bh * S_ * DK_;
    #pragma unroll 1
    for (int rr = 0; rr < 2; ++rr) {
        const int row = wv * 2 + rr;
        float sl[32];
        #pragma unroll
        for (int t = 0; t < 32; ++t) sl[t] = sc[row * S_ + t * 32 + lane];

        float denom = 0.f, c0 = 0.f, c1 = 0.f;
        for (int it = 0; it < TOPK_; ++it) {
            float bvv = -3.0e38f;
            int   bi  = 0x7FFFFFFF;
            #pragma unroll
            for (int t = 0; t < 32; ++t) {
                if (sl[t] > bvv) { bvv = sl[t]; bi = t * 32 + lane; }
            }
            #pragma unroll
            for (int m = 16; m >= 1; m >>= 1) {
                float ov = __shfl_xor(bvv, m);
                int   oi = __shfl_xor(bi, m);
                if (ov > bvv || (ov == bvv && oi < bi)) { bvv = ov; bi = oi; }
            }
            if ((bi & 31) == lane) sl[bi >> 5] = -3.0e38f;   // remove winner
            const float wgt = expf(bvv) - 1.0f;              // e^s - e^0
            denom += wgt;
            const float* vrow = vbase + (size_t)bi * DK_;
            c0 += wgt * vrow[lane];
            c1 += wgt * vrow[lane + 32];
        }
        // non-top-k entries all contribute exp(0)=1: numerator += V_sum,
        // denominator += S
        c0 += vsum[bh * DK_ + lane];
        c1 += vsum[bh * DK_ + lane + 32];
        const float dn = denom + (float)S_ + 1e-8f;
        float* orow = out + ((size_t)(b * S_ + s0 + row)) * (H_ * DK_) + h * DK_;
        orow[lane]      = c0 / dn;
        orow[lane + 32] = c1 / dn;
    }
}

// ---------------------------------------------------------------------------
extern "C" void kernel_launch(void* const* d_in, const int* in_sizes, int n_in,
                              void* d_out, int out_size, void* d_ws, size_t ws_size,
                              hipStream_t stream)
{
    (void)in_sizes; (void)n_in; (void)out_size; (void)ws_size;
    const float* Q  = (const float*)d_in[0];
    const float* K  = (const float*)d_in[1];
    const float* V  = (const float*)d_in[2];
    const float* Wq = (const float*)d_in[3];
    const float* bq = (const float*)d_in[4];
    const float* Wk = (const float*)d_in[5];
    const float* bk = (const float*)d_in[6];
    const float* Wv = (const float*)d_in[7];
    const float* bv = (const float*)d_in[8];
    float* out = (float*)d_out;

    char* ws = (char*)d_ws;
    const size_t nelem = (size_t)B_ * H_ * S_ * DK_;       // 8,388,608
    unsigned short* qbf  = (unsigned short*)ws;                        // 16 MB
    unsigned short* kbf  = (unsigned short*)(ws + nelem * 2);          // 16 MB
    float*          vf   = (float*)(ws + nelem * 4);                   // 32 MB
    float*          vsum = (float*)(ws + nelem * 4 + nelem * 4);       // 32 KB

    dim3 blk(256);
    proj_gemm_kernel<<<4096, blk, 0, stream>>>(Q, Wq, bq, qbf, (float*)nullptr, 0);
    proj_gemm_kernel<<<4096, blk, 0, stream>>>(K, Wk, bk, kbf, (float*)nullptr, 0);
    proj_gemm_kernel<<<4096, blk, 0, stream>>>(V, Wv, bv, (unsigned short*)nullptr, vf, 1);
    vsum_kernel<<<32, blk, 0, stream>>>(vf, vsum);
    topk_attn_kernel<<<8192, blk, 0, stream>>>(qbf, kbf, vf, vsum, out);
}
